// MultiHeadDiffAttention_71399536328796
// MI455X (gfx1250) — matrix operand order repaired and checked by reading, not compiled, not run on hardware
//
#include <hip/hip_runtime.h>

// ---------------------------------------------------------------------------
// Problem constants
// ---------------------------------------------------------------------------
#define BATCH 2
#define SEQ   2048
#define DMODEL 512
#define NHEAD 8
#define HDIM  64
#define NGROUP 4
#define LAMBDA_INIT 0.2f
#define GN_EPS 1e-5f
#define NROW (BATCH * SEQ)            // 4096

typedef __attribute__((ext_vector_type(16))) __bf16 v16bf;
typedef __attribute__((ext_vector_type(8)))  float  v8f;

union Frag {
    v16bf v;
    unsigned int u[8];
};

__device__ __forceinline__ v8f wmma_bf16(v16bf a, v16bf b, v8f c) {
    // D = A(16x32 bf16) * B(32x16 bf16) + C(16x16 f32)
    return __builtin_amdgcn_wmma_f32_16x16x32_bf16(
        false, a, false, b, (short)0, c, false, false);
}

__device__ __forceinline__ unsigned short f2bf(float x) {
    unsigned int u = __float_as_uint(x);
    unsigned int r = u + 0x7FFFu + ((u >> 16) & 1u);   // round-to-nearest-even
    return (unsigned short)(r >> 16);
}

// ---------------------------------------------------------------------------
// Elementwise f32 -> bf16 cast
// ---------------------------------------------------------------------------
__global__ void cast_bf16_kernel(const float* __restrict__ src,
                                 unsigned short* __restrict__ dst, int n) {
    int i = blockIdx.x * blockDim.x + threadIdx.x;
    if (i < n) dst[i] = f2bf(src[i]);
}

// ---------------------------------------------------------------------------
// lambda[h] = exp(lq1*lk1) - exp(lq2*lk2) + LAMBDA_INIT
// ---------------------------------------------------------------------------
__global__ void lambda_kernel(const float* __restrict__ lq1, const float* __restrict__ lk1,
                              const float* __restrict__ lq2, const float* __restrict__ lk2,
                              float* __restrict__ lam) {
    int h = threadIdx.x;
    if (h < NHEAD) {
        lam[h] = __expf(lq1[h] * lk1[h]) - __expf(lq2[h] * lk2[h]) + LAMBDA_INIT;
    }
}

// ---------------------------------------------------------------------------
// Fragment loader for the GEMM: A from X row (contiguous k = 16*hh + s order), B from W rows
// (packed K-pairs are contiguous dwords -> compiler emits global_load_b128).
// ---------------------------------------------------------------------------
__device__ __forceinline__ void load_frags(const unsigned short* __restrict__ Xrow,
                                           const unsigned short* __restrict__ Wb,
                                           int k0, int hh, Frag& a, Frag* b) {
#pragma unroll
    for (int r = 0; r < 8; ++r) {
        int k = k0 + 16 * hh + 2 * r;
        a.u[r] = *(const unsigned int*)(Xrow + k);
    }
#pragma unroll
    for (int j = 0; j < 4; ++j) {
#pragma unroll
        for (int r = 0; r < 8; ++r) {
            int kk = k0 + 16 * hh + 2 * r;
            b[j].u[r] = *(const unsigned int*)(Wb + j * 16 * DMODEL + kk);
        }
    }
}

// ---------------------------------------------------------------------------
// GEMM: Y[m,n] = sum_k X[m,k] * W[n,k] (+ bias[n]); X bf16 [M][512], W bf16 [512][512]
// mode 0: store bf16 per-head layout  [B][H][S][HDIM]
// mode 1: store bf16 head-transposed  [B][H][HDIM][S]   (for V)
// mode 2: store f32 flat [M][512] (+bias)               (final projection)
// Block = 128 threads (4 waves); wave computes a 16x64 tile; block = 64x64 tile.
// Double-buffered software pipeline: compute chunk k while chunk k+64 loads.
// __launch_bounds__(128,1): let the allocator keep both buffers in VGPRs.
// ---------------------------------------------------------------------------
__global__ void __launch_bounds__(128, 1)
proj_kernel(const unsigned short* __restrict__ X,
            const unsigned short* __restrict__ W,
            const float* __restrict__ bias,
            void* __restrict__ outp,
            int mode) {
    const int lane = threadIdx.x & 31;
    const int wv   = threadIdx.x >> 5;          // 0..3
    const int nb   = blockIdx.x & 7;            // DMODEL/64 = 8 column blocks
    const int mb   = blockIdx.x >> 3;
    const int m0   = mb * 64 + wv * 16;
    const int n0   = nb * 64;
    const int m    = lane & 15;
    const int hh   = lane >> 4;

    const unsigned short* Xrow = X + (long)(m0 + m) * DMODEL;
    const unsigned short* Wb   = W + (long)(n0 + m) * DMODEL;

    v8f acc[4];
#pragma unroll
    for (int j = 0; j < 4; ++j) acc[j] = (v8f){0.f,0.f,0.f,0.f,0.f,0.f,0.f,0.f};

    Frag aA, bA[4], aB, bB[4];
    load_frags(Xrow, Wb, 0,  hh, aA, bA);
    load_frags(Xrow, Wb, 32, hh, aB, bB);

#pragma unroll 1
    for (int k0 = 0; k0 < DMODEL - 64; k0 += 64) {
#pragma unroll
        for (int j = 0; j < 4; ++j) acc[j] = wmma_bf16(aA.v, bA[j].v, acc[j]);
        load_frags(Xrow, Wb, k0 + 64, hh, aA, bA);
#pragma unroll
        for (int j = 0; j < 4; ++j) acc[j] = wmma_bf16(aB.v, bB[j].v, acc[j]);
        load_frags(Xrow, Wb, k0 + 96, hh, aB, bB);
    }
#pragma unroll
    for (int j = 0; j < 4; ++j) acc[j] = wmma_bf16(aA.v, bA[j].v, acc[j]);
#pragma unroll
    for (int j = 0; j < 4; ++j) acc[j] = wmma_bf16(aB.v, bB[j].v, acc[j]);

    if (mode == 2) {
        float* O = (float*)outp;
#pragma unroll
        for (int j = 0; j < 4; ++j) {
#pragma unroll
            for (int r = 0; r < 8; ++r) {
                int row = m0 + r + 8 * hh;
                int c   = n0 + 16 * j + m;
                O[(long)row * DMODEL + c] = acc[j][r] + bias[c];
            }
        }
    } else {
        unsigned short* O = (unsigned short*)outp;
#pragma unroll
        for (int j = 0; j < 4; ++j) {
#pragma unroll
            for (int r = 0; r < 8; ++r) {
                int row = m0 + r + 8 * hh;
                int c   = n0 + 16 * j + m;
                int b_  = row >> 11;            // row / SEQ
                int s_  = row & (SEQ - 1);
                int h_  = c >> 6;
                int e_  = c & 63;
                float v = acc[j][r] + (bias ? bias[c] : 0.f);
                long idx = (mode == 0)
                    ? (((long)(b_ * NHEAD + h_) * SEQ + s_) * HDIM + e_)
                    : (((long)(b_ * NHEAD + h_) * HDIM + e_) * SEQ + s_);
                O[idx] = f2bf(v);
            }
        }
    }
}

// ---------------------------------------------------------------------------
// Flash attention (one softmax): O[b,s,h*64+e] = softmax(Q K^T / 8) V
// Q,K: bf16 [B*H][S][64]; Vt: bf16 [B*H][64][S]; O: f32 [B*S][512]
// Block = 128 (4 waves); wave handles one 16-query tile, loops 32-key blocks.
// V-fragment loads are issued before the softmax VALU work so they are in
// flight while exp/reductions run; next K block is prefetched into cache.
// ---------------------------------------------------------------------------
__global__ void __launch_bounds__(128, 1)
flash_kernel(const unsigned short* __restrict__ Q,
             const unsigned short* __restrict__ K,
             const unsigned short* __restrict__ Vt,
             float* __restrict__ O) {
    __shared__ unsigned short plds[4][16][32];   // per-wave P staging (1KB/wave)

    const int lane = threadIdx.x & 31;
    const int wv   = threadIdx.x >> 5;
    const int wid  = blockIdx.x * 4 + wv;
    const int bh   = wid >> 7;                   // / (SEQ/16)
    const int qt   = wid & 127;
    const int b_   = bh >> 3;
    const int h_   = bh & 7;
    const int m    = lane & 15;
    const int hh   = lane >> 4;

    const unsigned short* qp = Q  + ((long)bh * SEQ + qt * 16) * HDIM;
    const unsigned short* kp = K  + (long)bh * SEQ * HDIM;
    const unsigned short* vp = Vt + (long)bh * HDIM * SEQ;

    Frag qa[2];
#pragma unroll
    for (int f = 0; f < 2; ++f)
#pragma unroll
        for (int r = 0; r < 8; ++r) {
            int k = 32 * f + 16 * hh + 2 * r;
            qa[f].u[r] = *(const unsigned int*)(qp + m * HDIM + k);
        }

    v8f acc[4];
#pragma unroll
    for (int j = 0; j < 4; ++j) acc[j] = (v8f){0.f,0.f,0.f,0.f,0.f,0.f,0.f,0.f};
    float mrow[8], lrow[8];
#pragma unroll
    for (int r = 0; r < 8; ++r) { mrow[r] = -1e30f; lrow[r] = 0.f; }

    const float SC = 0.125f * 1.44269504f;       // (1/sqrt(64)) * log2(e)

#pragma unroll 1
    for (int kk0 = 0; kk0 < SEQ; kk0 += 32) {
        // ---- K fragments for both 16-key tiles, both 32-wide d-chunks ----
        Frag kb[2][2];
#pragma unroll
        for (int t = 0; t < 2; ++t)
#pragma unroll
            for (int f = 0; f < 2; ++f)
#pragma unroll
                for (int r = 0; r < 8; ++r) {
                    int d = 32 * f + 16 * hh + 2 * r;
                    kb[t][f].u[r] =
                        *(const unsigned int*)(kp + (kk0 + 16 * t + m) * HDIM + d);
                }

        v8f s0 = (v8f){0.f,0.f,0.f,0.f,0.f,0.f,0.f,0.f};
        v8f s1 = (v8f){0.f,0.f,0.f,0.f,0.f,0.f,0.f,0.f};
        s0 = wmma_bf16(qa[0].v, kb[0][0].v, s0);
        s0 = wmma_bf16(qa[1].v, kb[0][1].v, s0);
        s1 = wmma_bf16(qa[0].v, kb[1][0].v, s1);
        s1 = wmma_bf16(qa[1].v, kb[1][1].v, s1);

        // ---- issue V fragment loads early: independent of softmax ----
        Frag vb[4];
#pragma unroll
        for (int j = 0; j < 4; ++j)
#pragma unroll
            for (int r = 0; r < 8; ++r)
                vb[j].u[r] = *(const unsigned int*)
                    (vp + (16 * j + m) * SEQ + kk0 + 16 * hh + 2 * r);

        // ---- prefetch next key block into cache (global_prefetch_b8) ----
        __builtin_prefetch(kp + (kk0 + 32 + m) * HDIM, 0, 0);
        __builtin_prefetch(kp + (kk0 + 48 + m) * HDIM, 0, 0);

        // ---- streaming softmax over the 32 new keys ----
#pragma unroll
        for (int r = 0; r < 8; ++r) {
            float a0 = s0[r] * SC;
            float a1 = s1[r] * SC;
            float mx = fmaxf(a0, a1);
#pragma unroll
            for (int off = 1; off < 16; off <<= 1)
                mx = fmaxf(mx, __shfl_xor(mx, off, 16));
            float mn = fmaxf(mrow[r], mx);
            float al = exp2f(mrow[r] - mn);
            float p0 = exp2f(a0 - mn);
            float p1 = exp2f(a1 - mn);
            float ps = p0 + p1;
#pragma unroll
            for (int off = 1; off < 16; off <<= 1)
                ps += __shfl_xor(ps, off, 16);
            lrow[r] = lrow[r] * al + ps;
            mrow[r] = mn;
            acc[0][r] *= al; acc[1][r] *= al; acc[2][r] *= al; acc[3][r] *= al;
            plds[wv][r + 8 * hh][m]      = f2bf(p0);
            plds[wv][r + 8 * hh][16 + m] = f2bf(p1);
        }
        asm volatile("s_wait_dscnt 0" ::: "memory");

        // reload P in A-matrix layout (same wave, in-order LDS)
        Frag pf;
#pragma unroll
        for (int r = 0; r < 8; ++r) {
            int k = 16 * hh + 2 * r;
            pf.u[r] = *(const unsigned int*)&plds[wv][m][k];
        }
#pragma unroll
        for (int j = 0; j < 4; ++j)
            acc[j] = wmma_bf16(pf.v, vb[j].v, acc[j]);
    }

#pragma unroll
    for (int j = 0; j < 4; ++j)
#pragma unroll
        for (int r = 0; r < 8; ++r) {
            int row = qt * 16 + r + 8 * hh;
            O[((long)(b_ * SEQ + row)) * DMODEL + h_ * HDIM + 16 * j + m] =
                acc[j][r] / lrow[r];
        }
}

// ---------------------------------------------------------------------------
// GroupNorm stats over (128 channels x 2048 positions) of o = O1 - lam[h]*O2
// ---------------------------------------------------------------------------
__global__ void gn_stats_kernel(const float* __restrict__ O1, const float* __restrict__ O2,
                                const float* __restrict__ lam, float* __restrict__ stats) {
    __shared__ float ssum[256], ssq[256];
    const int bg = blockIdx.x;                  // 0..7
    const int b_ = bg >> 2, g_ = bg & 3;
    const int c0 = g_ * (DMODEL / NGROUP);
    float sum = 0.f, sq = 0.f;
    for (int i = threadIdx.x; i < SEQ * (DMODEL / NGROUP); i += 256) {
        int s_ = i >> 7;
        int c  = c0 + (i & 127);
        int h_ = c >> 6;
        long idx = ((long)(b_ * SEQ + s_)) * DMODEL + c;
        float v = O1[idx] - lam[h_] * O2[idx];
        sum += v; sq += v * v;
    }
    ssum[threadIdx.x] = sum; ssq[threadIdx.x] = sq;
    __syncthreads();
    for (int st = 128; st > 0; st >>= 1) {
        if (threadIdx.x < st) {
            ssum[threadIdx.x] += ssum[threadIdx.x + st];
            ssq[threadIdx.x]  += ssq[threadIdx.x + st];
        }
        __syncthreads();
    }
    if (threadIdx.x == 0) {
        const float N = (float)(SEQ * (DMODEL / NGROUP));
        float mean = ssum[0] / N;
        float var  = ssq[0] / N - mean * mean;
        stats[2 * bg]     = mean;
        stats[2 * bg + 1] = rsqrtf(var + GN_EPS);
    }
}

// ---------------------------------------------------------------------------
// GroupNorm apply + affine + cast to bf16
// ---------------------------------------------------------------------------
__global__ void gn_apply_kernel(const float* __restrict__ O1, const float* __restrict__ O2,
                                const float* __restrict__ lam, const float* __restrict__ stats,
                                const float* __restrict__ gw, const float* __restrict__ gb,
                                unsigned short* __restrict__ XN) {
    long idx = (long)blockIdx.x * blockDim.x + threadIdx.x;
    if (idx >= (long)NROW * DMODEL) return;
    int c   = (int)(idx & (DMODEL - 1));
    int row = (int)(idx >> 9);
    int b_  = row >> 11;
    int h_  = c >> 6;
    int g_  = c >> 7;
    int bg  = b_ * NGROUP + g_;
    float v  = O1[idx] - lam[h_] * O2[idx];
    float xn = (v - stats[2 * bg]) * stats[2 * bg + 1] * gw[c] + gb[c];
    XN[idx] = f2bf(xn);
}

// ---------------------------------------------------------------------------
// Host-side orchestration
// ---------------------------------------------------------------------------
extern "C" void kernel_launch(void* const* d_in, const int* in_sizes, int n_in,
                              void* d_out, int out_size, void* d_ws, size_t ws_size,
                              hipStream_t stream) {
    const float* x     = (const float*)d_in[0];
    const float* K1_w  = (const float*)d_in[1];
    const float* K1_b  = (const float*)d_in[2];
    const float* Q1_w  = (const float*)d_in[3];
    const float* K2_w  = (const float*)d_in[4];
    const float* K2_b  = (const float*)d_in[5];
    const float* Q2_w  = (const float*)d_in[6];
    const float* V_w   = (const float*)d_in[7];
    const float* lq1   = (const float*)d_in[8];
    const float* lk1   = (const float*)d_in[9];
    const float* lq2   = (const float*)d_in[10];
    const float* lk2   = (const float*)d_in[11];
    const float* gn_w  = (const float*)d_in[12];
    const float* gn_b  = (const float*)d_in[13];
    const float* out_w = (const float*)d_in[14];
    const float* out_b = (const float*)d_in[15];

    char* ws = (char*)d_ws;
    const size_t SZ_XBF = (size_t)NROW * DMODEL * 2;             // 4 MiB
    const size_t SZ_W   = (size_t)DMODEL * DMODEL * 2;           // 512 KiB each
    const size_t SZ_HB  = (size_t)NROW * DMODEL * 2;             // per-head bf16 buf
    const size_t SZ_O   = (size_t)NROW * DMODEL * 4;             // fp32

    unsigned short* XBF = (unsigned short*)(ws);
    unsigned short* WQ1 = (unsigned short*)(ws + SZ_XBF + 0 * SZ_W);
    unsigned short* WK1 = (unsigned short*)(ws + SZ_XBF + 1 * SZ_W);
    unsigned short* WQ2 = (unsigned short*)(ws + SZ_XBF + 2 * SZ_W);
    unsigned short* WK2 = (unsigned short*)(ws + SZ_XBF + 3 * SZ_W);
    unsigned short* WV  = (unsigned short*)(ws + SZ_XBF + 4 * SZ_W);
    unsigned short* WO  = (unsigned short*)(ws + SZ_XBF + 5 * SZ_W);
    char* p = ws + SZ_XBF + 6 * SZ_W;
    unsigned short* Q1B = (unsigned short*)(p); p += SZ_HB;
    unsigned short* K1B = (unsigned short*)(p); p += SZ_HB;
    unsigned short* Q2B = (unsigned short*)(p); p += SZ_HB;
    unsigned short* K2B = (unsigned short*)(p); p += SZ_HB;
    unsigned short* VTB = (unsigned short*)(p); p += SZ_HB;
    float* O1  = (float*)(p); p += SZ_O;
    float* O2  = (float*)(p); p += SZ_O;
    unsigned short* XNB = (unsigned short*)(p); p += SZ_HB;
    float* LAM   = (float*)(p); p += 64;
    float* STATS = (float*)(p);

    const int nx = NROW * DMODEL;       // 2,097,152
    const int nw = DMODEL * DMODEL;     //   262,144

    cast_bf16_kernel<<<(nx + 255) / 256, 256, 0, stream>>>(x, XBF, nx);
    cast_bf16_kernel<<<(nw + 255) / 256, 256, 0, stream>>>(Q1_w,  WQ1, nw);
    cast_bf16_kernel<<<(nw + 255) / 256, 256, 0, stream>>>(K1_w,  WK1, nw);
    cast_bf16_kernel<<<(nw + 255) / 256, 256, 0, stream>>>(Q2_w,  WQ2, nw);
    cast_bf16_kernel<<<(nw + 255) / 256, 256, 0, stream>>>(K2_w,  WK2, nw);
    cast_bf16_kernel<<<(nw + 255) / 256, 256, 0, stream>>>(V_w,   WV,  nw);
    cast_bf16_kernel<<<(nw + 255) / 256, 256, 0, stream>>>(out_w, WO,  nw);
    lambda_kernel<<<1, 32, 0, stream>>>(lq1, lk1, lq2, lk2, LAM);

    const dim3 pgrid((NROW / 64) * (DMODEL / 64));  // 512 blocks
    proj_kernel<<<pgrid, 128, 0, stream>>>(XBF, WQ1, nullptr, Q1B, 0);
    proj_kernel<<<pgrid, 128, 0, stream>>>(XBF, WK1, K1_b,    K1B, 0);
    proj_kernel<<<pgrid, 128, 0, stream>>>(XBF, WQ2, nullptr, Q2B, 0);
    proj_kernel<<<pgrid, 128, 0, stream>>>(XBF, WK2, K2_b,    K2B, 0);
    proj_kernel<<<pgrid, 128, 0, stream>>>(XBF, WV,  nullptr, VTB, 1);

    const dim3 agrid(BATCH * NHEAD * (SEQ / 16) / 4);  // 512 blocks, 4 waves each
    flash_kernel<<<agrid, 128, 0, stream>>>(Q1B, K1B, VTB, O1);
    flash_kernel<<<agrid, 128, 0, stream>>>(Q2B, K2B, VTB, O2);

    gn_stats_kernel<<<BATCH * NGROUP, 256, 0, stream>>>(O1, O2, LAM, STATS);
    gn_apply_kernel<<<(nx + 255) / 256, 256, 0, stream>>>(O1, O2, LAM, STATS, gn_w, gn_b, XNB);

    proj_kernel<<<pgrid, 128, 0, stream>>>(XNB, WO, out_b, (float*)d_out, 2);
}
